// EHR_RNN_19464791786022
// MI455X (gfx1250) — compile-verified, matrix operand order checked by reading, not run on hardware
//
#include <hip/hip_runtime.h>
#include <hip/hip_bf16.h>
#include <stdint.h>

#define B_ 256
#define V_ 96
#define L_ 24
#define E_ 256
#define H_ 512
#define G4H_ (4 * H_)

typedef __attribute__((ext_vector_type(16))) _Float16 v16h;
typedef __attribute__((ext_vector_type(8)))  float    v8f;

union FragU { v16h v; uint32_t u[8]; };

// Load a 16x32 f16 fragment (A-layout; also valid as B when the [N,K] matrix is
// row-major, since W row-major == B^T) from a row-major matrix (global or LDS).
// Layout (ISA 7.12.2, 16-bit A 16x32):
//   lanes 0-15 : row = lane,    VGPR j<4: K = 2j,2j+1 ; VGPR j>=4: K = 16+2(j-4)..
//   lanes16-31 : row = lane-16, VGPR j<4: K = 8+2j    ; VGPR j>=4: K = 24+2(j-4)..
__device__ __forceinline__ v16h load_frag16(const _Float16* __restrict__ base,
                                            int row0, int k0, int ldk, int lane) {
    const int r    = lane & 15;
    const int kpo  = (lane & 16) ? 4 : 0;   // pair offset (K/2)
    const uint32_t* p = (const uint32_t*)(base + (size_t)(row0 + r) * ldk + k0);
    FragU f;
#pragma unroll
    for (int j = 0; j < 4; ++j) f.u[j]     = p[kpo + j];
#pragma unroll
    for (int j = 0; j < 4; ++j) f.u[4 + j] = p[8 + kpo + j];
    return f.v;
}

// ---------------------------------------------------------------- prep -------
__global__ __launch_bounds__(256) void prep_kernel(
    const float* __restrict__ Wih, const float* __restrict__ Whh,
    const float* __restrict__ bih, const float* __restrict__ bhh,
    _Float16* __restrict__ Wih16, _Float16* __restrict__ Whh16,
    float* __restrict__ bias, _Float16* __restrict__ h0, float* __restrict__ c0) {
    const int i      = blockIdx.x * blockDim.x + threadIdx.x;
    const int stride = gridDim.x * blockDim.x;
    for (int k = i; k < G4H_ * E_; k += stride) Wih16[k] = (_Float16)Wih[k];
    for (int k = i; k < G4H_ * H_; k += stride) Whh16[k] = (_Float16)Whh[k];
    for (int k = i; k < G4H_;      k += stride) bias[k]  = bih[k] + bhh[k];
    for (int k = i; k < B_ * H_;   k += stride) { h0[k] = (_Float16)0.0f; c0[k] = 0.0f; }
}

// ------------------------------------------------------- embedding bag -------
// One block per (b,v); thread e handles one embedding dim. Output time-major
// X[v][b][e] in f16, zeroed for padded (invalid) timesteps.
__global__ __launch_bounds__(256) void embed_kernel(
    const int* __restrict__ codes, const int* __restrict__ n_visits,
    const float* __restrict__ emb, _Float16* __restrict__ X) {
    const int idx = blockIdx.x;          // b*V + v
    const int b   = idx / V_;
    const int v   = idx - b * V_;
    const int e   = threadIdx.x;
    float acc = 0.0f;
    if (v >= V_ - n_visits[b]) {
        const int* cp = codes + (size_t)idx * L_;
#pragma unroll
        for (int l = 0; l < L_; ++l)
            acc += emb[(size_t)cp[l] * E_ + e];
    }
    X[((size_t)v * B_ + b) * E_ + e] = (_Float16)acc;
}

// ------------------------------------------------------------ LSTM step ------
// Block = 8 waves, all sharing one hidden tile nt; wave w owns batch tile
// m0 = ((blockIdx&1)*8 + w)*16. W_hh panel for nt (4 gates x 16 rows x 512 K,
// 64 KB) is async-DMA'd into LDS once per step, then WMMA B-fragments come
// from ds_load_b128. g = X[t]@Wih^T + h@Whh^T + bias -> cell update.
__global__ __launch_bounds__(256) void lstm_step_kernel(
    const _Float16* __restrict__ X,     // [V,B,E]
    const _Float16* __restrict__ Wih,   // [4H,E] row-major == B^T
    const _Float16* __restrict__ Whh,   // [4H,H] row-major == B^T
    const float*    __restrict__ bias,  // [4H] (b_ih + b_hh)
    const _Float16* __restrict__ h_in,  // [B,H]
    _Float16*       __restrict__ h_out, // [B,H]
    float*          __restrict__ h32,   // [B,H]
    float*          __restrict__ c,     // [B,H] in/out
    int t) {
    __shared__ _Float16 sWhh[4 * 16 * H_];   // 64 KB: [gate][16 rows][H_ K]

    const int nt    = blockIdx.x >> 1;       // 0..31
    const int mhalf = blockIdx.x & 1;
    const int wid   = threadIdx.x >> 5;
    const int lane  = threadIdx.x & 31;
    const int n0    = nt * 16;
    const int m0    = (mhalf * 8 + wid) * 16;

    // ---- stage W_hh panel into LDS with async global->LDS DMA ----
    // Rows g*H+n0 .. +16 are contiguous (full K rows), so each gate panel is
    // one flat 16*H_ halves chunk. 16B per lane per issue, 16 issues/thread.
#pragma unroll
    for (int g = 0; g < 4; ++g) {
        const _Float16* gp = Whh + ((size_t)g * H_ + n0) * H_;
        _Float16*       sp = sWhh + g * 16 * H_;
        for (int i = threadIdx.x; i < (16 * H_) / 8; i += 256) {
            const _Float16* ga = gp + (size_t)i * 8;
            unsigned so = (unsigned)(size_t)(sp + (size_t)i * 8);  // LDS byte offset
            asm volatile("global_load_async_to_lds_b128 %0, %1, off"
                         :: "v"(so), "v"(ga) : "memory");
        }
    }
    asm volatile("s_wait_asynccnt 0" ::: "memory");
    __syncthreads();

    v8f acc0 = {}, acc1 = {}, acc2 = {}, acc3 = {};

    // input contribution: K over E (B-fragments from global; small, L2/WGP$ hot)
    const _Float16* Xt = X + (size_t)t * B_ * E_;
    for (int k0 = 0; k0 < E_; k0 += 32) {
        v16h a  = load_frag16(Xt, m0, k0, E_, lane);
        v16h b0 = load_frag16(Wih, 0 * H_ + n0, k0, E_, lane);
        v16h b1 = load_frag16(Wih, 1 * H_ + n0, k0, E_, lane);
        v16h b2 = load_frag16(Wih, 2 * H_ + n0, k0, E_, lane);
        v16h b3 = load_frag16(Wih, 3 * H_ + n0, k0, E_, lane);
        acc0 = __builtin_amdgcn_wmma_f32_16x16x32_f16(false, a, false, b0, (short)0, acc0, false, false);
        acc1 = __builtin_amdgcn_wmma_f32_16x16x32_f16(false, a, false, b1, (short)0, acc1, false, false);
        acc2 = __builtin_amdgcn_wmma_f32_16x16x32_f16(false, a, false, b2, (short)0, acc2, false, false);
        acc3 = __builtin_amdgcn_wmma_f32_16x16x32_f16(false, a, false, b3, (short)0, acc3, false, false);
    }
    // recurrent contribution: K over H (B-fragments from LDS)
    for (int k0 = 0; k0 < H_; k0 += 32) {
        v16h a  = load_frag16(h_in, m0, k0, H_, lane);
        v16h b0 = load_frag16(sWhh, 0 * 16, k0, H_, lane);
        v16h b1 = load_frag16(sWhh, 1 * 16, k0, H_, lane);
        v16h b2 = load_frag16(sWhh, 2 * 16, k0, H_, lane);
        v16h b3 = load_frag16(sWhh, 3 * 16, k0, H_, lane);
        acc0 = __builtin_amdgcn_wmma_f32_16x16x32_f16(false, a, false, b0, (short)0, acc0, false, false);
        acc1 = __builtin_amdgcn_wmma_f32_16x16x32_f16(false, a, false, b1, (short)0, acc1, false, false);
        acc2 = __builtin_amdgcn_wmma_f32_16x16x32_f16(false, a, false, b2, (short)0, acc2, false, false);
        acc3 = __builtin_amdgcn_wmma_f32_16x16x32_f16(false, a, false, b3, (short)0, acc3, false, false);
    }

    // elementwise cell update. C/D layout: lane l, reg r -> m = (l<16? r : 8+r),
    // n = l & 15.
    const int j  = n0 + (lane & 15);
    const float bi = bias[0 * H_ + j];
    const float bf = bias[1 * H_ + j];
    const float bg = bias[2 * H_ + j];
    const float bo = bias[3 * H_ + j];
#pragma unroll
    for (int r = 0; r < 8; ++r) {
        const int m  = (lane < 16) ? r : (8 + r);
        const size_t idx = (size_t)(m0 + m) * H_ + j;
        const float gi = 1.0f / (1.0f + __expf(-(acc0[r] + bi)));
        const float gf = 1.0f / (1.0f + __expf(-(acc1[r] + bf)));
        const float gg = tanhf(acc2[r] + bg);
        const float go = 1.0f / (1.0f + __expf(-(acc3[r] + bo)));
        const float cn = gf * c[idx] + gi * gg;
        c[idx] = cn;
        const float hn = go * tanhf(cn);
        h32[idx]   = hn;
        h_out[idx] = (_Float16)hn;
    }
}

// -------------------------------------------------------------- output -------
__global__ __launch_bounds__(256) void out_kernel(
    const float* __restrict__ h, const float* __restrict__ W_out,
    const float* __restrict__ b_out, float* __restrict__ out) {
    const int b = threadIdx.x;           // B_ == 256 threads
    float acc = b_out[0];
    for (int jj = 0; jj < H_; ++jj)
        acc += h[(size_t)b * H_ + jj] * W_out[jj];
    out[b] = 1.0f / (1.0f + __expf(-acc));
}

// ------------------------------------------------------------- launcher ------
extern "C" void kernel_launch(void* const* d_in, const int* in_sizes, int n_in,
                              void* d_out, int out_size, void* d_ws, size_t ws_size,
                              hipStream_t stream) {
    const int*   codes    = (const int*)d_in[0];
    const int*   n_visits = (const int*)d_in[1];
    const float* emb      = (const float*)d_in[2];
    const float* W_ih     = (const float*)d_in[3];
    const float* W_hh     = (const float*)d_in[4];
    const float* b_ih     = (const float*)d_in[5];
    const float* b_hh     = (const float*)d_in[6];
    const float* W_out    = (const float*)d_in[7];
    const float* b_out    = (const float*)d_in[8];
    float*       out      = (float*)d_out;

    char* ws = (char*)d_ws;
    size_t off = 0;
    auto alloc = [&](size_t bytes) -> void* {
        void* p = ws + off;
        off += (bytes + 255) & ~(size_t)255;
        return p;
    };
    _Float16* X     = (_Float16*)alloc((size_t)V_ * B_ * E_ * 2);   // 12.6 MB
    _Float16* Wih16 = (_Float16*)alloc((size_t)G4H_ * E_ * 2);      //  1.0 MB
    _Float16* Whh16 = (_Float16*)alloc((size_t)G4H_ * H_ * 2);      //  2.1 MB
    float*    bias  = (float*)alloc((size_t)G4H_ * 4);
    _Float16* hA    = (_Float16*)alloc((size_t)B_ * H_ * 2);
    _Float16* hB    = (_Float16*)alloc((size_t)B_ * H_ * 2);
    float*    h32   = (float*)alloc((size_t)B_ * H_ * 4);
    float*    c     = (float*)alloc((size_t)B_ * H_ * 4);

    prep_kernel<<<512, 256, 0, stream>>>(W_ih, W_hh, b_ih, b_hh,
                                         Wih16, Whh16, bias, hA, c);
    embed_kernel<<<B_ * V_, 256, 0, stream>>>(codes, n_visits, emb, X);

    _Float16* hin = hA;
    _Float16* hout = hB;
    for (int t = 0; t < V_; ++t) {
        // 64 blocks x 8 waves; each block = one hidden tile (x2 batch halves)
        lstm_step_kernel<<<64, 256, 0, stream>>>(X, Wih16, Whh16, bias,
                                                 hin, hout, h32, c, t);
        _Float16* tmp = hin; hin = hout; hout = tmp;
    }
    out_kernel<<<1, 256, 0, stream>>>(h32, W_out, b_out, out);
}
